// SingleHeadSelfAttention_15272903705142
// MI455X (gfx1250) — matrix operand order repaired and checked by reading, not compiled, not run on hardware
//
#include <hip/hip_runtime.h>

typedef __attribute__((ext_vector_type(16))) __bf16 v16bf;
typedef __attribute__((ext_vector_type(8)))  __bf16 v8bf;
typedef __attribute__((ext_vector_type(4)))  __bf16 v4bf;
typedef __attribute__((ext_vector_type(8)))  float  v8f;

#define B_  8
#define T_  2048
#define C_  1024
#define H_  1024
#define QR_ 32
#define SCALE_ 0.03125f   // 1/sqrt(1024)

__device__ __forceinline__ v16bf ld16(const __bf16* p) {
  v16bf r;
  ((v8bf*)&r)[0] = *(const v8bf*)p;
  ((v8bf*)&r)[1] = *(const v8bf*)(p + 8);
  return r;
}

// CDNA5 async global->LDS copy (16B per lane, tracked by ASYNCcnt).
// Generic pointers to LDS carry the LDS offset in their low 32 bits.
__device__ __forceinline__ void async_b128(const void* g, void* l) {
  asm volatile("global_load_async_to_lds_b128 %0, %1, off"
               :: "v"((unsigned)(unsigned long long)l),
                  "v"((unsigned long long)g)
               : "memory");
}
__device__ __forceinline__ void wait_async0() {
  asm volatile("s_wait_asynccnt 0" ::: "memory");
}

// ---------------------------------------------------------------------------
// Phase 0a: convert x to bf16 (read once, feed phase 1 from L2 at half width)
// ---------------------------------------------------------------------------
__global__ __launch_bounds__(256) void cvt_x(
    const float* __restrict__ x, __bf16* __restrict__ xb)
{
  size_t i = ((size_t)blockIdx.x * 256 + threadIdx.x) * 8;
  const float4 a = *(const float4*)(x + i);
  const float4 b = *(const float4*)(x + i + 4);
  v8bf o;
  o[0] = (__bf16)a.x; o[1] = (__bf16)a.y; o[2] = (__bf16)a.z; o[3] = (__bf16)a.w;
  o[4] = (__bf16)b.x; o[5] = (__bf16)b.y; o[6] = (__bf16)b.z; o[7] = (__bf16)b.w;
  *(v8bf*)(xb + i) = o;
}

// ---------------------------------------------------------------------------
// Phase 0b: convert + transpose W[C][H] -> Wt[H][C] bf16 (LDS tile transpose)
// ---------------------------------------------------------------------------
__global__ __launch_bounds__(256) void cvt_w_t(
    const float* __restrict__ Wq, const float* __restrict__ Wk,
    const float* __restrict__ Wv,
    __bf16* __restrict__ Wtq, __bf16* __restrict__ Wtk, __bf16* __restrict__ Wtv)
{
  __shared__ __attribute__((aligned(16))) __bf16 Tt[64][64];
  const int tid = threadIdx.x;
  const int kb = blockIdx.x * 64, nb = blockIdx.y * 64, wz = blockIdx.z;
  const float* W = (wz == 0) ? Wq : (wz == 1) ? Wk : Wv;
  __bf16* Wt     = (wz == 0) ? Wtq : (wz == 1) ? Wtk : Wtv;

  #pragma unroll
  for (int i = 0; i < 4; ++i) {
    int idx = tid + i * 256;                   // 1024 float4 chunks
    int kr = idx >> 4, c = (idx & 15) * 4;
    const float4 v = *(const float4*)&W[(size_t)(kb + kr) * H_ + nb + c];
    Tt[c + 0][kr] = (__bf16)v.x; Tt[c + 1][kr] = (__bf16)v.y;
    Tt[c + 2][kr] = (__bf16)v.z; Tt[c + 3][kr] = (__bf16)v.w;
  }
  __syncthreads();
  #pragma unroll
  for (int i = 0; i < 2; ++i) {
    int idx = tid + i * 256;                   // 512 v8bf chunks
    int n = idx >> 3, k8 = (idx & 7) * 8;
    *(v8bf*)&Wt[(size_t)(nb + n) * C_ + kb + k8] = *(const v8bf*)&Tt[n][k8];
  }
}

// ---------------------------------------------------------------------------
// Phase 1: QKV projections. Y = xb @ Wt^T, all-bf16 operands.
// Workgroup tile 128x64, K-step 128, DOUBLE-BUFFERED async staging:
// issue tile i+1 DMA, compute tile i. 16 WMMAs per barrier, one barrier/tile.
// ---------------------------------------------------------------------------
__global__ __launch_bounds__(256) void qkv_proj(
    const __bf16* __restrict__ xb,
    const __bf16* __restrict__ Wtq, const __bf16* __restrict__ Wtk,
    const __bf16* __restrict__ Wtv,
    __bf16* __restrict__ Qb, __bf16* __restrict__ Kb, __bf16* __restrict__ Vt)
{
  __shared__ __attribute__((aligned(16))) __bf16 As[2][128][128]; // 64 KB
  __shared__ __attribute__((aligned(16))) __bf16 Bs[2][64][128];  // 32 KB

  const int tid  = threadIdx.x, lane = tid & 31, wv = tid >> 5;
  const int ml   = lane & 15,   half = lane >> 4;
  const int nb   = blockIdx.x * 64;
  const int mb   = blockIdx.y * 128;
  const int wz   = blockIdx.z;
  const __bf16* Wt = (wz == 0) ? Wtq : (wz == 1) ? Wtk : Wtv;

  v8f acc[4] = {};

  auto stage = [&](int kk, int buf) {
    #pragma unroll
    for (int i = 0; i < 8; ++i) {            // x tile 128x128: 2048 chunks
      int idx = tid + i * 256;
      int r = idx >> 4, c8 = (idx & 15) * 8;
      async_b128(xb + (size_t)(mb + r) * C_ + kk + c8, &As[buf][r][c8]);
    }
    #pragma unroll
    for (int i = 0; i < 4; ++i) {            // Wt tile 64x128: 1024 chunks
      int idx = tid + i * 256;
      int n = idx >> 4, k8 = (idx & 15) * 8;
      async_b128(Wt + (size_t)(nb + n) * C_ + kk + k8, &Bs[buf][n][k8]);
    }
  };

  stage(0, 0);
  for (int it = 0; it < C_ / 128; ++it) {
    wait_async0();         // own loads for tile `it` have landed
    __syncthreads();       // publish; also: nobody still reads buf (it+1)&1
    if (it + 1 < C_ / 128) stage((it + 1) * 128, (it + 1) & 1);  // overlap DMA
    const int buf = it & 1;

    #pragma unroll
    for (int ks = 0; ks < 128; ks += 32) {
      v16bf a, b0, b1, b2, b3;
      ((v8bf*)&a)[0] = *(const v8bf*)&As[buf][wv * 16 + ml][ks + 16 * half];
      ((v8bf*)&a)[1] = *(const v8bf*)&As[buf][wv * 16 + ml][ks + 16 * half + 8];
      b0 = ld16(&Bs[buf][ 0 + ml][ks + 16 * half]);
      b1 = ld16(&Bs[buf][16 + ml][ks + 16 * half]);
      b2 = ld16(&Bs[buf][32 + ml][ks + 16 * half]);
      b3 = ld16(&Bs[buf][48 + ml][ks + 16 * half]);
      acc[0] = __builtin_amdgcn_wmma_f32_16x16x32_bf16(false, a, false, b0, (short)0, acc[0], false, false);
      acc[1] = __builtin_amdgcn_wmma_f32_16x16x32_bf16(false, a, false, b1, (short)0, acc[1], false, false);
      acc[2] = __builtin_amdgcn_wmma_f32_16x16x32_bf16(false, a, false, b2, (short)0, acc[2], false, false);
      acc[3] = __builtin_amdgcn_wmma_f32_16x16x32_bf16(false, a, false, b3, (short)0, acc[3], false, false);
    }
  }

  // Epilogue. C/D layout: VGPR v -> M = v + 8*half, N = lane%16
  if (wz == 2) {
    // per lane & nt, the 8 acc values are CONSECUTIVE t in one Vt row -> b128 store
    const int m0 = mb + wv * 16 + 8 * half;
    const int bi = m0 >> 11, t = m0 & (T_ - 1);
    #pragma unroll
    for (int nt = 0; nt < 4; ++nt) {
      v8bf o;
      #pragma unroll
      for (int v = 0; v < 8; ++v) o[v] = (__bf16)acc[nt][v];
      int n = nb + nt * 16 + ml;
      *(v8bf*)&Vt[((size_t)bi * H_ + n) * T_ + t] = o;
    }
  } else {
    __bf16* __restrict__ dst = (wz == 0) ? Qb : Kb;
    const float sc = (wz == 0) ? SCALE_ : 1.0f;
    #pragma unroll
    for (int nt = 0; nt < 4; ++nt) {
      #pragma unroll
      for (int v = 0; v < 8; ++v) {
        int m = mb + wv * 16 + v + 8 * half;
        int n = nb + nt * 16 + ml;
        dst[(size_t)m * H_ + n] = (__bf16)(acc[nt][v] * sc);
      }
    }
  }
}

// ---------------------------------------------------------------------------
// Phase 2: flash-style causal attention. One workgroup (8 waves) per 32-query
// tile. Per 128-key block: each wave computes two 16x16 score tiles (shared
// B fragment) over H=1024, online softmax in LDS, then each wave accumulates
// P@V for its private 128-wide H slice (two M-tiles share each V fragment).
// ---------------------------------------------------------------------------
__global__ __launch_bounds__(256) void attn_fwd(
    const __bf16* __restrict__ Qb, const __bf16* __restrict__ Kb,
    const __bf16* __restrict__ Vt, float* __restrict__ out)
{
  __shared__ __attribute__((aligned(16))) __bf16 Qs[QR_][H_];   // 64 KB
  __shared__ __attribute__((aligned(16))) float  Ss[QR_][128];  // 16 KB
  __shared__ __attribute__((aligned(16))) __bf16 Ps[QR_][128];  // 8 KB
  __shared__ float m_s[QR_], l_s[QR_], alpha_s[QR_];

  const int tid = threadIdx.x, lane = tid & 31, wv = tid >> 5;
  const int ml  = lane & 15,   half = lane >> 4;
  const int b   = blockIdx.x >> 6;          // T/32 = 64 q-tiles per batch
  const int q0  = (blockIdx.x & 63) << 5;
  const int h0  = wv * 128;                 // this wave's output H slice

  // async-load the 32x1024 Q tile (4096 16B chunks, 16/thread)
  const __bf16* qsrc = Qb + ((size_t)b * T_ + q0) * H_;
  #pragma unroll
  for (int i = 0; i < 16; ++i) {
    int idx = (tid + i * 256) * 8;
    int r = idx >> 10, c = idx & 1023;
    async_b128(&qsrc[(size_t)r * H_ + c], &Qs[r][c]);
  }
  if (tid < QR_) { m_s[tid] = -__builtin_inff(); l_s[tid] = 0.f; }
  wait_async0();
  __syncthreads();

  v8f acc[2][8] = {};

  for (int kb = 0; kb < q0 + QR_; kb += 128) {
    // warm V rows of this key block toward the caches for the PV stage
    #pragma unroll
    for (int nt = 0; nt < 8; ++nt)
      __builtin_prefetch(Vt + ((size_t)b * H_ + h0 + nt * 16 + ml) * T_ + kb, 0, 0);

    // ---- scores: wave wv owns key tile [kb + 16*wv, +16) for BOTH m-tiles
    const int key0 = kb + wv * 16;
    v8f s0 = {}, s1 = {};
    const __bf16* krow = Kb + ((size_t)b * T_ + key0 + ml) * H_ + 16 * half;
    #pragma unroll 2
    for (int hh = 0; hh < H_; hh += 32) {
      v16bf bb = ld16(krow + hh);           // contiguous row of K = B fragment
      v16bf a0, a1;
      ((v8bf*)&a0)[0] = *(const v8bf*)&Qs[ml][hh + 16 * half];
      ((v8bf*)&a0)[1] = *(const v8bf*)&Qs[ml][hh + 16 * half + 8];
      ((v8bf*)&a1)[0] = *(const v8bf*)&Qs[16 + ml][hh + 16 * half];
      ((v8bf*)&a1)[1] = *(const v8bf*)&Qs[16 + ml][hh + 16 * half + 8];
      s0 = __builtin_amdgcn_wmma_f32_16x16x32_bf16(
          false, a0, false, bb, (short)0, s0, false, false);
      s1 = __builtin_amdgcn_wmma_f32_16x16x32_bf16(
          false, a1, false, bb, (short)0, s1, false, false);
    }
    const int kglob = key0 + ml;
    #pragma unroll
    for (int v = 0; v < 8; ++v) {
      int mrow = v + 8 * half;
      Ss[mrow][wv * 16 + ml]      = (kglob <= q0 + mrow)      ? s0[v] : -__builtin_inff();
      Ss[16 + mrow][wv * 16 + ml] = (kglob <= q0 + 16 + mrow) ? s1[v] : -__builtin_inff();
    }
    __syncthreads();

    // ---- online softmax: wave wv handles rows 4wv..4wv+3 (128 cols each)
    #pragma unroll
    for (int rr = 0; rr < 4; ++rr) {
      const int r = wv * 4 + rr;
      const float4 v4 = *(const float4*)&Ss[r][lane * 4];
      float vs[4] = { v4.x, v4.y, v4.z, v4.w };
      float vmax = fmaxf(fmaxf(vs[0], vs[1]), fmaxf(vs[2], vs[3]));
      #pragma unroll
      for (int off = 16; off; off >>= 1) vmax = fmaxf(vmax, __shfl_xor(vmax, off, 32));
      const float mold  = m_s[r];
      const float mnew  = fmaxf(mold, vmax);
      const float alpha = __expf(mold - mnew);   // first block: exp(-inf)=0
      float psum = 0.f;
      v4bf p4;
      #pragma unroll
      for (int j = 0; j < 4; ++j) {
        float p = __expf(vs[j] - mnew);
        p4[j] = (__bf16)p;
        psum += p;
      }
      *(v4bf*)&Ps[r][lane * 4] = p4;             // packed 8-byte LDS store
      #pragma unroll
      for (int off = 16; off; off >>= 1) psum += __shfl_xor(psum, off, 32);
      if (lane == 0) { m_s[r] = mnew; alpha_s[r] = alpha; l_s[r] = l_s[r] * alpha + psum; }
    }
    __syncthreads();

    // ---- rescale running output, then acc += P @ V for this wave's H slice
    #pragma unroll
    for (int v = 0; v < 8; ++v) {
      const float al0 = alpha_s[v + 8 * half];
      const float al1 = alpha_s[16 + v + 8 * half];
      #pragma unroll
      for (int nt = 0; nt < 8; ++nt) { acc[0][nt][v] *= al0; acc[1][nt][v] *= al1; }
    }
    const __bf16* vbase = Vt + ((size_t)b * H_ + h0 + ml) * T_ + kb + 16 * half;
    #pragma unroll
    for (int kk = 0; kk < 128; kk += 32) {
      v16bf a0, a1;
      ((v8bf*)&a0)[0] = *(const v8bf*)&Ps[ml][kk + 16 * half];
      ((v8bf*)&a0)[1] = *(const v8bf*)&Ps[ml][kk + 16 * half + 8];
      ((v8bf*)&a1)[0] = *(const v8bf*)&Ps[16 + ml][kk + 16 * half];
      ((v8bf*)&a1)[1] = *(const v8bf*)&Ps[16 + ml][kk + 16 * half + 8];
      #pragma unroll
      for (int nt = 0; nt < 8; nt += 2) {
        v16bf bb0 = ld16(vbase + (size_t)(nt    ) * 16 * T_ + kk);
        v16bf bb1 = ld16(vbase + (size_t)(nt + 1) * 16 * T_ + kk);
        acc[0][nt]     = __builtin_amdgcn_wmma_f32_16x16x32_bf16(false, a0, false, bb0, (short)0, acc[0][nt],     false, false);
        acc[1][nt]     = __builtin_amdgcn_wmma_f32_16x16x32_bf16(false, a1, false, bb0, (short)0, acc[1][nt],     false, false);
        acc[0][nt + 1] = __builtin_amdgcn_wmma_f32_16x16x32_bf16(false, a0, false, bb1, (short)0, acc[0][nt + 1], false, false);
        acc[1][nt + 1] = __builtin_amdgcn_wmma_f32_16x16x32_bf16(false, a1, false, bb1, (short)0, acc[1][nt + 1], false, false);
      }
    }
  }

  __syncthreads();
  #pragma unroll
  for (int mt = 0; mt < 2; ++mt) {
    #pragma unroll
    for (int nt = 0; nt < 8; ++nt) {
      #pragma unroll
      for (int v = 0; v < 8; ++v) {
        int m = mt * 16 + v + 8 * half;
        out[((size_t)b * T_ + q0 + m) * H_ + h0 + nt * 16 + ml] =
            acc[mt][nt][v] / l_s[m];
      }
    }
  }
}

// ---------------------------------------------------------------------------
extern "C" void kernel_launch(void* const* d_in, const int* in_sizes, int n_in,
                              void* d_out, int out_size, void* d_ws, size_t ws_size,
                              hipStream_t stream) {
  const float* x  = (const float*)d_in[0];
  const float* Wq = (const float*)d_in[1];
  const float* Wk = (const float*)d_in[2];
  const float* Wv = (const float*)d_in[3];

  // workspace: xb 32MB | Wt q/k/v 3x2MB | Qb/Kb/Vt 3x32MB  = 134 MB
  __bf16* xb  = (__bf16*)d_ws;
  __bf16* Wtq = xb  + (size_t)B_ * T_ * C_;
  __bf16* Wtk = Wtq + (size_t)C_ * H_;
  __bf16* Wtv = Wtk + (size_t)C_ * H_;
  __bf16* Qb  = Wtv + (size_t)C_ * H_;
  __bf16* Kb  = Qb  + (size_t)B_ * T_ * H_;
  __bf16* Vt  = Kb  + (size_t)B_ * T_ * H_;

  cvt_x<<<(B_ * T_ * C_) / (256 * 8), 256, 0, stream>>>(x, xb);
  dim3 g0(C_ / 64, H_ / 64, 3);
  cvt_w_t<<<g0, 256, 0, stream>>>(Wq, Wk, Wv, Wtq, Wtk, Wtv);

  dim3 g1(H_ / 64, (B_ * T_) / 128, 3);
  qkv_proj<<<g1, 256, 0, stream>>>(xb, Wtq, Wtk, Wtv, Qb, Kb, Vt);
  attn_fwd<<<(B_ * T_) / QR_, 256, 0, stream>>>(Qb, Kb, Vt, (float*)d_out);
}
